// GraphormerLayer_56530359550884
// MI455X (gfx1250) — matrix operand order repaired and checked by reading, not compiled, not run on hardware
//
#include <hip/hip_runtime.h>
#include <hip/hip_bf16.h>
#include <math.h>
#include <stdint.h>

typedef _Float16 half_t;
typedef __attribute__((ext_vector_type(4)))  _Float16 v4h;
typedef __attribute__((ext_vector_type(8)))  _Float16 v8h;
typedef __attribute__((ext_vector_type(16))) _Float16 v16h;
typedef __attribute__((ext_vector_type(8)))  float    v8f;
typedef __attribute__((ext_vector_type(4)))  unsigned int v4u;
typedef __attribute__((ext_vector_type(8)))  int      v8i;
typedef __attribute__((ext_vector_type(4)))  int      v4i;

// ---------------------------------------------------------------- helpers

__device__ inline v8f vzero8() {
  v8f z;
#pragma unroll
  for (int i = 0; i < 8; ++i) z[i] = 0.0f;
  return z;
}

// Build a 16-half fragment from two 8-half chunks (8B-granular LDS/global loads).
__device__ inline v16h ld_frag(const half_t* p0, const half_t* p1) {
  v4h a0 = *(const v4h*)(p0);
  v4h a1 = *(const v4h*)(p0 + 4);
  v4h b0 = *(const v4h*)(p1);
  v4h b1 = *(const v4h*)(p1 + 4);
  v16h f;
#pragma unroll
  for (int i = 0; i < 4; ++i) {
    f[i] = a0[i]; f[4 + i] = a1[i]; f[8 + i] = b0[i]; f[12 + i] = b1[i];
  }
  return f;
}

__device__ inline v8f wmma_f16(v16h a, v16h b, v8f c) {
  return __builtin_amdgcn_wmma_f32_16x16x32_f16(
      false, a, false, b, (short)0, c, false, false);
}

// Async global -> LDS copy, 16 bytes per lane (ASYNCcnt-tracked DMA path).
__device__ inline void async_copy_b128(unsigned lds_off, const void* gaddr) {
  asm volatile("global_load_async_to_lds_b128 %0, %1, off"
               :: "v"(lds_off), "v"((unsigned long long)(uintptr_t)gaddr)
               : "memory");
}
__device__ inline void wait_asynccnt0() {
  asm volatile("s_wait_asynccnt 0x0" ::: "memory");
}

// TDM: DMA a 32-row x 512-half tile (row-major, contiguous rows) into LDS with a
// 16B pad inserted after every 1024B row (LDS row stride 520 halves).
// Issued by a single wave; tracked with TENSORcnt.
__device__ inline void tdm_load_k_tile(const half_t* gsrc, unsigned lds_off) {
  unsigned long long ga = (unsigned long long)(uintptr_t)gsrc;
  v4u g0;
  g0[0] = 1u;                                   // count=1 user descriptor
  g0[1] = lds_off;                              // lds_addr (bytes)
  g0[2] = (unsigned)ga;                         // global_addr[31:0]
  g0[3] = (unsigned)(ga >> 32) | (2u << 30);    // global_addr[56:32] | type=2
  v8i g1;
  g1[0] = (1 << 16)        // data_size = 2 bytes
        | (1 << 20)        // pad_enable
        | (7 << 22)        // pad_interval: 256 dwords (= one 1024B row)
        | (3 << 25);       // pad_amount: 4 dwords (= 16B)
  g1[1] = (512 << 16);     // tensor_dim0 = 512
  g1[2] = (32 << 16);      // tensor_dim1 = 32
  g1[3] = (512 << 16);     // tile_dim0 = 512
  g1[4] = 32;              // tile_dim1 = 32
  g1[5] = 512;             // tensor_dim0_stride = 512 elements
  g1[6] = 0;
  g1[7] = 0;
  v4i zz4;
#pragma unroll
  for (int i = 0; i < 4; ++i) zz4[i] = 0;
  v8i zz8;
#pragma unroll
  for (int i = 0; i < 8; ++i) zz8[i] = 0;
  // clang-23 / therock-10.0 signature: (g0, g1, g2, g3, g4, cpol)
  __builtin_amdgcn_tensor_load_to_lds(g0, g1, zz4, zz4, zz8, 0);
}

// ---------------------------------------------------------------- f32->f16

__global__ __launch_bounds__(256) void cvt_f32_f16(
    const float* __restrict__ in, half_t* __restrict__ out, int n) {
  int i = blockIdx.x * 256 + threadIdx.x;
  if (i < n) out[i] = (half_t)in[i];
}

// ---------------------------------------------------------------- LayerNorm (row per block, D=512)

__global__ __launch_bounds__(256) void ln_fused(
    const float* __restrict__ x, const float* __restrict__ g,
    const float* __restrict__ b, half_t* __restrict__ out) {
  __shared__ float red[256];
  __shared__ float red2[256];
  const int row = blockIdx.x, t = threadIdx.x;
  float a0 = x[(size_t)row * 512 + t];
  float a1 = x[(size_t)row * 512 + 256 + t];
  red[t] = a0 + a1;
  red2[t] = a0 * a0 + a1 * a1;
  __syncthreads();
#pragma unroll
  for (int s = 128; s > 0; s >>= 1) {
    if (t < s) { red[t] += red[t + s]; red2[t] += red2[t + s]; }
    __syncthreads();
  }
  float mean = red[0] * (1.0f / 512.0f);
  float var  = red2[0] * (1.0f / 512.0f) - mean * mean;
  float rinv = rsqrtf(var + 1e-5f);
  out[(size_t)row * 512 + t]       = (half_t)((a0 - mean) * rinv * g[t] + b[t]);
  out[(size_t)row * 512 + 256 + t] = (half_t)((a1 - mean) * rinv * g[256 + t] + b[256 + t]);
}

// ---------------------------------------------------------------- tiled WMMA GEMM
// C[M x Nc] = A[M x K](f16) * B[K x Nc](f16) + bias, M fixed 4096.
// Block: 256 threads (8 waves), tile 128x64; wave = 32x32 (2x2 WMMA tiles); K-step 32.
// Double-buffered LDS; A tile staged via async global->LDS DMA (overlaps WMMA).
// EPI: 0 = store f16, 1 = exact GELU -> f16, 2 = residual(f32) + result -> f32.

#define A_LD 72   // halves; 144B rows: 16B-aligned for async B128, bank-conflict-free
#define B_LD 40

template <int EPI>
__global__ __launch_bounds__(256) void gemm16(
    const half_t* __restrict__ A, const half_t* __restrict__ B,
    const float* __restrict__ bias, const float* __restrict__ resid,
    half_t* __restrict__ outH, float* __restrict__ outF, int K, int Nc) {
  __shared__ half_t Al[2][128 * A_LD];
  __shared__ half_t Btl[2][64 * B_LD];

  const int tid = threadIdx.x;
  const int lane = tid & 31, w = tid >> 5;
  const int hf = lane >> 4, r = lane & 15;
  const int rowBase = blockIdx.y * 128;
  const int colBase = blockIdx.x * 64;
  const int wr = (w & 3) * 32, wc = (w >> 2) * 32;

  v8f acc[2][2];
#pragma unroll
  for (int i = 0; i < 2; ++i)
#pragma unroll
    for (int j = 0; j < 2; ++j) acc[i][j] = vzero8();

  auto loadA = [&](int kb, int buf) {
#pragma unroll
    for (int it = 0; it < 2; ++it) {
      int i = tid + it * 256;
      int rr = i >> 2, cc = i & 3;
      async_copy_b128((unsigned)(uintptr_t)&Al[buf][rr * A_LD + cc * 8],
                      A + (size_t)(rowBase + rr) * K + kb + cc * 8);
    }
  };
  auto loadB = [&](int kb, int buf) {
    int kk = tid >> 3, cs = tid & 7;
    v8h valb = *(const v8h*)(B + (size_t)(kb + kk) * Nc + colBase + cs * 8);
#pragma unroll
    for (int j = 0; j < 8; ++j) Btl[buf][(cs * 8 + j) * B_LD + kk] = valb[j];
  };

  loadA(0, 0);
  loadB(0, 0);
  wait_asynccnt0();
  __syncthreads();

  int buf = 0;
  for (int kb = 0; kb < K; kb += 32) {
    if (kb + 32 < K) {           // kick off next tile's DMA behind the math
      loadA(kb + 32, buf ^ 1);
      loadB(kb + 32, buf ^ 1);
    }

    const half_t* a0p = &Al[buf][(wr + r) * A_LD + hf * 8];
    const half_t* a1p = &Al[buf][(wr + 16 + r) * A_LD + hf * 8];
    v16h a0 = ld_frag(a0p, a0p + 16);
    v16h a1 = ld_frag(a1p, a1p + 16);
    const half_t* b0p = &Btl[buf][(wc + r) * B_LD + hf * 8];
    const half_t* b1p = &Btl[buf][(wc + 16 + r) * B_LD + hf * 8];
    v16h b0 = ld_frag(b0p, b0p + 16);
    v16h b1 = ld_frag(b1p, b1p + 16);

    acc[0][0] = wmma_f16(a0, b0, acc[0][0]);
    acc[0][1] = wmma_f16(a0, b1, acc[0][1]);
    acc[1][0] = wmma_f16(a1, b0, acc[1][0]);
    acc[1][1] = wmma_f16(a1, b1, acc[1][1]);

    wait_asynccnt0();
    __syncthreads();
    buf ^= 1;
  }

  // --- epilogue: C layout row = v + hf*8, col = lane&15 ---
#pragma unroll
  for (int j = 0; j < 2; ++j) {
    int col = colBase + wc + j * 16 + r;
    float bv = bias[col];
#pragma unroll
    for (int i = 0; i < 2; ++i) {
#pragma unroll
      for (int v = 0; v < 8; ++v) {
        int row = rowBase + wr + i * 16 + hf * 8 + v;
        float val = acc[i][j][v] + bv;
        if (EPI == 0) {
          outH[(size_t)row * Nc + col] = (half_t)val;
        } else if (EPI == 1) {
          float ge = 0.5f * val * (1.0f + erff(val * 0.70710678118654752f));
          outH[(size_t)row * Nc + col] = (half_t)ge;
        } else {
          outF[(size_t)row * Nc + col] = resid[(size_t)row * Nc + col] + val;
        }
      }
    }
  }
}

// ---------------------------------------------------------------- flash attention
// Block: 256 threads = 8 waves; wave w = head w; block covers 16 query rows for ALL
// heads so the (n,m,h)-layout bias tile is read with fully dense cachelines.
// K tile staged by the Tensor Data Mover (with 16B/row LDS pad); V transposed and
// bias staged by VALU; next tiles prefetched into L2 during the math.

#define KL_LD 520            // 512 + 8-halves pad (matches TDM pad descriptor)
#define VT_LD 40             // 32 + 8 pad
#define PL_LD 40
#define KL_BYTES (32 * KL_LD * 2)          // 33280
#define VT_BYTES (512 * VT_LD * 2)         // 40960
#define BL_BYTES (16 * 256 * 4)            // 16384
#define PL_BYTES (8 * 16 * PL_LD * 2)      // 10240
#define FLASH_LDS (KL_BYTES + VT_BYTES + BL_BYTES + PL_BYTES)

__global__ __launch_bounds__(256) void flash_attn(
    const half_t* __restrict__ q, const half_t* __restrict__ k,
    const half_t* __restrict__ v, const float* __restrict__ bias,
    half_t* __restrict__ ctxo) {
  extern __shared__ char smem[];
  half_t* Kl = (half_t*)smem;                                    // [32][KL_LD]
  half_t* Vt = (half_t*)(smem + KL_BYTES);                       // [512][VT_LD] (d, m)
  float*  Bl = (float*)(smem + KL_BYTES + VT_BYTES);             // [16][32*8]
  half_t* Pl = (half_t*)(smem + KL_BYTES + VT_BYTES + BL_BYTES); // [8][16][PL_LD]

  const int tid = threadIdx.x;
  const int lane = tid & 31, h = tid >> 5;       // wave == head
  const int hf = lane >> 4, r = lane & 15;
  const int qb = blockIdx.x * 16;

  // Q fragments for this head (16 rows x 64 d), A-layout, loaded once.
  const half_t* qrow = q + (size_t)(qb + r) * 512 + h * 64;
  v16h aq0 = ld_frag(qrow + hf * 8, qrow + 16 + hf * 8);        // d 0..31
  v16h aq1 = ld_frag(qrow + 32 + hf * 8, qrow + 48 + hf * 8);   // d 32..63

  v8f ctx[4];
  float runm[8], runs[8];
#pragma unroll
  for (int dt = 0; dt < 4; ++dt) ctx[dt] = vzero8();
#pragma unroll
  for (int i = 0; i < 8; ++i) { runm[i] = -1e30f; runs[i] = 0.0f; }

  for (int mb = 0; mb < 4096; mb += 32) {
    // ---- stage K tile via TDM (wave 0 issues the descriptor) ----
    if (h == 0) {
      tdm_load_k_tile(k + (size_t)mb * 512, (unsigned)(uintptr_t)Kl);
    }
    // ---- stage V (transposed) and bias cooperatively ----
#pragma unroll
    for (int it = 0; it < 8; ++it) {
      int i = tid + it * 256;
      int row = i >> 6, cc = i & 63;
      v8h vv = *(const v8h*)(v + (size_t)(mb + row) * 512 + cc * 8);
#pragma unroll
      for (int j = 0; j < 8; ++j) Vt[(cc * 8 + j) * VT_LD + row] = vv[j];
    }
#pragma unroll
    for (int it = 0; it < 16; ++it) {
      int i = tid + it * 256;
      int n = i >> 8, mh = i & 255;   // mh = m*8 + h, contiguous in memory
      Bl[i] = bias[(size_t)(qb + n) * 32768 + (size_t)mb * 8 + mh];
    }
    // prefetch next iteration's K/V/bias tiles into L2 while we compute
    if (mb + 32 < 4096) {
      __builtin_prefetch(k + (size_t)(mb + 32) * 512 + tid * 64, 0, 1);
      __builtin_prefetch(v + (size_t)(mb + 32) * 512 + tid * 64, 0, 1);
      if (tid < 128)
        __builtin_prefetch(bias + (size_t)(qb + (tid >> 3)) * 32768 +
                               (size_t)(mb + 32) * 8 + (tid & 7) * 32, 0, 1);
    }
    if (h == 0) __builtin_amdgcn_s_wait_tensorcnt(0);
    __syncthreads();

    // ---- S = Q K^T / sqrt(DH) + bias : two 16x16 tiles ----
    float s_t[2][8];
#pragma unroll
    for (int mt = 0; mt < 2; ++mt) {
      const half_t* kb0 = Kl + (mt * 16 + r) * KL_LD + h * 64;  // lane = m column
      v16h bk0 = ld_frag(kb0 + hf * 8, kb0 + 16 + hf * 8);        // d 0..31
      v16h bk1 = ld_frag(kb0 + 32 + hf * 8, kb0 + 48 + hf * 8); // d 32..63
      v8f s = vzero8();
      s = wmma_f16(aq0, bk0, s);
      s = wmma_f16(aq1, bk1, s);
#pragma unroll
      for (int vv2 = 0; vv2 < 8; ++vv2)
        s_t[mt][vv2] = s[vv2] * 0.125f +
                       Bl[(vv2 + hf * 8) * 256 + (mt * 16 + r) * 8 + h];
    }

    // ---- online softmax (row values live across the 16 lanes of each half) ----
#pragma unroll
    for (int vv2 = 0; vv2 < 8; ++vv2) {
      float mx = fmaxf(s_t[0][vv2], s_t[1][vv2]);
      mx = fmaxf(mx, __shfl_xor(mx, 1));
      mx = fmaxf(mx, __shfl_xor(mx, 2));
      mx = fmaxf(mx, __shfl_xor(mx, 4));
      mx = fmaxf(mx, __shfl_xor(mx, 8));
      float nm = fmaxf(runm[vv2], mx);
      float al = __expf(runm[vv2] - nm);
      float e0 = __expf(s_t[0][vv2] - nm);
      float e1 = __expf(s_t[1][vv2] - nm);
      runs[vv2] = runs[vv2] * al + e0 + e1;
      runm[vv2] = nm;
#pragma unroll
      for (int dt = 0; dt < 4; ++dt) ctx[dt][vv2] *= al;
      // stage P (C-layout -> row-major) for re-fragmentation as A-matrix
      Pl[h * (16 * PL_LD) + (vv2 + hf * 8) * PL_LD + r]      = (half_t)e0;
      Pl[h * (16 * PL_LD) + (vv2 + hf * 8) * PL_LD + 16 + r] = (half_t)e1;
    }
    asm volatile("s_wait_dscnt 0x0" ::: "memory");

    // ---- ctx += P(16x32) @ V(32x64) ----
    const half_t* pp = Pl + h * (16 * PL_LD) + r * PL_LD + hf * 8;
    v16h ap = ld_frag(pp, pp + 16);
#pragma unroll
    for (int dt = 0; dt < 4; ++dt) {
      const half_t* vb = Vt + (h * 64 + dt * 16 + r) * VT_LD + hf * 8; // lane = d col
      v16h bv = ld_frag(vb, vb + 16);
      ctx[dt] = wmma_f16(ap, bv, ctx[dt]);
    }
    __syncthreads();
  }

  // ---- finalize: divide by softmax denominator, store ctx as f16 ----
#pragma unroll
  for (int vv2 = 0; vv2 < 8; ++vv2) {
    float s = runs[vv2];
    s += __shfl_xor(s, 1);
    s += __shfl_xor(s, 2);
    s += __shfl_xor(s, 4);
    s += __shfl_xor(s, 8);
    float inv = 1.0f / s;
#pragma unroll
    for (int dt = 0; dt < 4; ++dt)
      ctxo[(size_t)(qb + vv2 + hf * 8) * 512 + h * 64 + dt * 16 + r] =
          (half_t)(ctx[dt][vv2] * inv);
  }
}

// ---------------------------------------------------------------- launch

extern "C" void kernel_launch(void* const* d_in, const int* in_sizes, int n_in,
                              void* d_out, int out_size, void* d_ws, size_t ws_size,
                              hipStream_t stream) {
  const float* x    = (const float*)d_in[0];
  const float* abia = (const float*)d_in[1];
  const float* ln1g = (const float*)d_in[2];
  const float* ln1b = (const float*)d_in[3];
  const float* wq = (const float*)d_in[4];
  const float* bq = (const float*)d_in[5];
  const float* wk = (const float*)d_in[6];
  const float* bk = (const float*)d_in[7];
  const float* wv = (const float*)d_in[8];
  const float* bv = (const float*)d_in[9];
  const float* wo = (const float*)d_in[10];
  const float* bo = (const float*)d_in[11];
  const float* ln2g = (const float*)d_in[12];
  const float* ln2b = (const float*)d_in[13];
  const float* w1 = (const float*)d_in[14];
  const float* b1 = (const float*)d_in[15];
  const float* w2 = (const float*)d_in[16];
  const float* b2 = (const float*)d_in[17];
  float* out = (float*)d_out;

  char* ws = (char*)d_ws;
  size_t off = 0;
  auto alloc = [&](size_t bytes) {
    char* p = ws + off;
    off += (bytes + 255) & ~(size_t)255;
    return p;
  };
  half_t* wqh  = (half_t*)alloc(512 * 512 * 2);
  half_t* wkh  = (half_t*)alloc(512 * 512 * 2);
  half_t* wvh  = (half_t*)alloc(512 * 512 * 2);
  half_t* woh  = (half_t*)alloc(512 * 512 * 2);
  half_t* w1h  = (half_t*)alloc(512 * 2048 * 2);
  half_t* w2h  = (half_t*)alloc(2048 * 512 * 2);
  half_t* xn1h = (half_t*)alloc((size_t)4096 * 512 * 2);
  half_t* qh   = (half_t*)alloc((size_t)4096 * 512 * 2);
  half_t* kh   = (half_t*)alloc((size_t)4096 * 512 * 2);
  half_t* vh   = (half_t*)alloc((size_t)4096 * 512 * 2);
  half_t* ctxh = (half_t*)alloc((size_t)4096 * 512 * 2);
  float*  xat  = (float*)alloc((size_t)4096 * 512 * 4);
  half_t* xn2h = (half_t*)alloc((size_t)4096 * 512 * 2);
  half_t* hh   = (half_t*)alloc((size_t)4096 * 2048 * 2);

  // weights -> f16
  cvt_f32_f16<<<512 * 512 / 256, 256, 0, stream>>>(wq, wqh, 512 * 512);
  cvt_f32_f16<<<512 * 512 / 256, 256, 0, stream>>>(wk, wkh, 512 * 512);
  cvt_f32_f16<<<512 * 512 / 256, 256, 0, stream>>>(wv, wvh, 512 * 512);
  cvt_f32_f16<<<512 * 512 / 256, 256, 0, stream>>>(wo, woh, 512 * 512);
  cvt_f32_f16<<<512 * 2048 / 256, 256, 0, stream>>>(w1, w1h, 512 * 2048);
  cvt_f32_f16<<<512 * 2048 / 256, 256, 0, stream>>>(w2, w2h, 2048 * 512);

  // LN1
  ln_fused<<<4096, 256, 0, stream>>>(x, ln1g, ln1b, xn1h);

  // QKV projections (M=4096, K=512, Nc=512)
  gemm16<0><<<dim3(8, 32), 256, 0, stream>>>(xn1h, wqh, bq, nullptr, qh, nullptr, 512, 512);
  gemm16<0><<<dim3(8, 32), 256, 0, stream>>>(xn1h, wkh, bk, nullptr, kh, nullptr, 512, 512);
  gemm16<0><<<dim3(8, 32), 256, 0, stream>>>(xn1h, wvh, bv, nullptr, vh, nullptr, 512, 512);

  // attention core (bias read once, dense cachelines; K/V restream from L2)
  flash_attn<<<256, 256, FLASH_LDS, stream>>>(qh, kh, vh, abia, ctxh);

  // O-projection + residual -> f32
  gemm16<2><<<dim3(8, 32), 256, 0, stream>>>(ctxh, woh, bo, x, nullptr, xat, 512, 512);

  // LN2
  ln_fused<<<4096, 256, 0, stream>>>(xat, ln2g, ln2b, xn2h);

  // FFN1 + exact GELU (K=512, Nc=2048)
  gemm16<1><<<dim3(32, 32), 256, 0, stream>>>(xn2h, w1h, b1, nullptr, hh, nullptr, 512, 2048);

  // FFN2 + residual -> d_out (K=2048, Nc=512)
  gemm16<2><<<dim3(8, 32), 256, 0, stream>>>(hh, w2h, b2, xat, nullptr, out, 2048, 512);

  (void)in_sizes; (void)n_in; (void)out_size; (void)ws_size;
}
